// Capsule_9491877724636
// MI455X (gfx1250) — matrix operand order repaired and checked by reading, not compiled, not run on hardware
//
#include <hip/hip_runtime.h>
#include <stdint.h>

#define B_BATCH  64
#define SEQ      512
#define DIN      512
#define NCAP     32
#define DCAP     64
#define NOUT     (NCAP*DCAP)   // 2048
#define ROUTINGS 3
#define EPSF     1e-7f

typedef __attribute__((ext_vector_type(16))) __bf16 v16bf;
typedef __attribute__((ext_vector_type(8)))  __bf16 v8bf;
typedef __attribute__((ext_vector_type(8)))  float  v8f;

typedef unsigned short bf16raw;
typedef unsigned int   u32;

__device__ __forceinline__ bf16raw f32_to_bf16(float f) {
  union { float f; u32 u; } v; v.f = f;
  u32 u = v.u;
  u32 r = u + 0x7FFFu + ((u >> 16) & 1u);   // round-to-nearest-even
  return (bf16raw)(r >> 16);
}
__device__ __forceinline__ float bf16_to_f32(bf16raw h) {
  union { u32 u; float f; } v; v.u = ((u32)h) << 16;
  return v.f;
}

// ---------------------------------------------------------------------------
// Kernel 1: convert u_vecs fp32 -> bf16 (row-major (B*S, Din))
// ---------------------------------------------------------------------------
__global__ void cvt_A(const float* __restrict__ X, bf16raw* __restrict__ Y, int n) {
  int i = blockIdx.x * blockDim.x + threadIdx.x;
  if (i < n) Y[i] = f32_to_bf16(X[i]);
}

// ---------------------------------------------------------------------------
// Kernel 2: W (Din x Nout, row-major) -> Wt bf16 (Nout x Din, row-major)
// so every WMMA B-fragment lane reads 16 contiguous K values.
// ---------------------------------------------------------------------------
__global__ void cvt_Wt(const float* __restrict__ W, bf16raw* __restrict__ Wt) {
  int i = blockIdx.x * blockDim.x + threadIdx.x;   // i < NOUT*DIN
  int n = i >> 9;        // / DIN
  int k = i & 511;       // % DIN
  Wt[i] = f32_to_bf16(W[(size_t)k * NOUT + n]);
}

// ---------------------------------------------------------------------------
// Kernel 3: GEMM u_hat = u_vecs @ W via v_wmma_f32_16x16x32_bf16.
// Block = 256 threads = 8 waves. Wave computes a 16(M) x 64(N) tile.
// blockIdx.x: M tile group (128 rows), blockIdx.y: capsule (64 cols).
// Output written in (B, N, S, D) bf16 layout directly.
// ---------------------------------------------------------------------------
__global__ void __launch_bounds__(256)
gemm_uhat(const bf16raw* __restrict__ Abf,
          const bf16raw* __restrict__ Wt,
          bf16raw* __restrict__ Uh) {
  const int lane = threadIdx.x & 31;
  const int wave = threadIdx.x >> 5;
  const int m0   = blockIdx.x * 128 + wave * 16;   // global row base of tile
  const int cap  = blockIdx.y;                     // capsule index
  const int n0   = cap * DCAP;                     // global col base (64 cols)

  const int row  = lane & 15;                      // A row / B col / D col
  const int half = lane >> 4;                      // lane group
  const int ka   = half * 16;                      // A K sub-offset
  const int kb   = half * 16;                      // B K sub-offset

  const bf16raw* Arow = Abf + (size_t)(m0 + row) * DIN;
  const bf16raw* Bcol = Wt  + (size_t)(n0 + row) * DIN + kb;

  v8f acc0 = {}, acc1 = {}, acc2 = {}, acc3 = {};

  for (int k0 = 0; k0 < DIN; k0 += 32) {
    // A fragment 16x32 bf16: two 16B chunks per lane, 16 contiguous K per lane (matches B)
    v8bf alo = *(const v8bf*)(Arow + k0 + ka);
    v8bf ahi = *(const v8bf*)(Arow + k0 + ka + 8);
    v16bf a  = __builtin_shufflevector(alo, ahi,
                 0,1,2,3,4,5,6,7,8,9,10,11,12,13,14,15);

    // B fragments 32x16 bf16: 16 contiguous K per lane (32B aligned)
    v16bf b0 = *(const v16bf*)(Bcol + k0 + 0 * 16 * DIN);
    v16bf b1 = *(const v16bf*)(Bcol + k0 + 1 * 16 * DIN);
    v16bf b2 = *(const v16bf*)(Bcol + k0 + 2 * 16 * DIN);
    v16bf b3 = *(const v16bf*)(Bcol + k0 + 3 * 16 * DIN);

    acc0 = __builtin_amdgcn_wmma_f32_16x16x32_bf16(false, a, false, b0, (short)0, acc0, false, false);
    acc1 = __builtin_amdgcn_wmma_f32_16x16x32_bf16(false, a, false, b1, (short)0, acc1, false, false);
    acc2 = __builtin_amdgcn_wmma_f32_16x16x32_bf16(false, a, false, b2, (short)0, acc2, false, false);
    acc3 = __builtin_amdgcn_wmma_f32_16x16x32_bf16(false, a, false, b3, (short)0, acc3, false, false);
  }

  // D layout: element i -> M = i + 8*half, N = lane&15. Store into (B,N,S,D).
  const int bb = m0 >> 9;             // batch
  const int s0 = (m0 & 511) + half * 8;
  bf16raw* Ub = Uh + (size_t)(bb * NCAP + cap) * SEQ * DCAP;
#pragma unroll
  for (int i = 0; i < 8; ++i) {
    size_t base = (size_t)(s0 + i) * DCAP + row;
    Ub[base + 0 * 16] = f32_to_bf16(acc0[i]);
    Ub[base + 1 * 16] = f32_to_bf16(acc1[i]);
    Ub[base + 2 * 16] = f32_to_bf16(acc2[i]);
    Ub[base + 3 * 16] = f32_to_bf16(acc3[i]);
  }
}

// ---------------------------------------------------------------------------
// Kernel 4: dynamic routing. One workgroup per batch, 512 threads = 16 waves.
// b-logits (32x512 f32 = 64KB) + outputs (32x64) live in LDS (320KB/WGP).
// ---------------------------------------------------------------------------
__global__ void __launch_bounds__(512)
routing(const bf16raw* __restrict__ Uh, float* __restrict__ Out) {
  __shared__ float blog[NCAP * SEQ];   // b logits, overwritten by c in place
  __shared__ float outs[NCAP * DCAP];  // capsule outputs
  __shared__ float invn[NCAP];         // 1/sqrt(||v||^2+eps)

  const int t    = threadIdx.x;
  const int bb   = blockIdx.x;
  const int lane = t & 31;
  const int wave = t >> 5;
  const bf16raw* Ub = Uh + (size_t)bb * NCAP * SEQ * DCAP;

  for (int i = t; i < NCAP * SEQ; i += 512) blog[i] = 0.f;
  __syncthreads();

  for (int it = 0; it < ROUTINGS; ++it) {
    // --- softmax over N per s: thread t <-> s = t (purely per-thread) ---
    {
      const int s = t;
      float vals[NCAP];
      float mx = -1e30f;
#pragma unroll
      for (int n = 0; n < NCAP; ++n) {
        float v = blog[n * SEQ + s];
        vals[n] = v;
        mx = fmaxf(mx, v);
      }
      float sum = 0.f;
#pragma unroll
      for (int n = 0; n < NCAP; ++n) {
        float e = __expf(vals[n] - mx);
        vals[n] = e;
        sum += e;
      }
      float inv = 1.f / sum;
#pragma unroll
      for (int n = 0; n < NCAP; ++n) blog[n * SEQ + s] = vals[n] * inv;
    }
    __syncthreads();

    // --- outputs[n][d] = sum_s c[n][s] * uh[n][s][d] ---
    // thread: d = t&63 (lanes -> contiguous d: coalesced bf16 reads),
    //         n in {t>>6, +8, +16, +24}; c[n][s] is wave-uniform LDS read.
    {
      const int d  = t & 63;
      const int nb = t >> 6;
#pragma unroll
      for (int j = 0; j < 4; ++j) {
        const int n = nb + j * 8;
        const bf16raw* up = Ub + (size_t)n * SEQ * DCAP + d;
        const float*   cp = blog + n * SEQ;
        float acc = 0.f;
        for (int s = 0; s < SEQ; ++s)
          acc += cp[s] * bf16_to_f32(up[(size_t)s * DCAP]);
        outs[n * DCAP + d] = acc;
      }
    }
    __syncthreads();

    // --- squash: v / sqrt(||v||^2 + eps) ---
    if (t < NCAP) {
      float ss = 0.f;
#pragma unroll
      for (int d = 0; d < DCAP; ++d) {
        float v = outs[t * DCAP + d];
        ss += v * v;
      }
      invn[t] = rsqrtf(ss + EPSF);
    }
    __syncthreads();
    for (int i = t; i < NCAP * DCAP; i += 512) outs[i] *= invn[i >> 6];
    __syncthreads();

    // --- agreement logits: b[n][s] = sum_d outs[n][d]*uh[n][s][d] ---
    if (it < ROUTINGS - 1) {
      for (int p = wave; p < NCAP * SEQ; p += 16) {
        const int n = p >> 9;        // p / 512
        const int s = p & 511;       // p % 512
        const bf16raw* up = Ub + ((size_t)n * SEQ + s) * DCAP;
        float part = outs[n * DCAP + lane]      * bf16_to_f32(up[lane])
                   + outs[n * DCAP + lane + 32] * bf16_to_f32(up[lane + 32]);
#pragma unroll
        for (int off = 16; off > 0; off >>= 1)
          part += __shfl_xor(part, off, 32);
        if (lane == 0) blog[n * SEQ + s] = part;
      }
      __syncthreads();
    }
  }

  // --- write final outputs (B, N, D) fp32 ---
  for (int i = t; i < NCAP * DCAP; i += 512)
    Out[(size_t)bb * NCAP * DCAP + i] = outs[i];
}

// ---------------------------------------------------------------------------
// Host launcher
// ---------------------------------------------------------------------------
extern "C" void kernel_launch(void* const* d_in, const int* in_sizes, int n_in,
                              void* d_out, int out_size, void* d_ws, size_t ws_size,
                              hipStream_t stream) {
  const float* u_vecs = (const float*)d_in[0];   // (64, 512, 512) fp32
  const float* W      = (const float*)d_in[1];   // (512, 2048)    fp32
  float* Out = (float*)d_out;                    // (64, 32, 64)   fp32

  char* ws = (char*)d_ws;
  const size_t nA  = (size_t)B_BATCH * SEQ * DIN;     // 16,777,216
  const size_t nWt = (size_t)NOUT * DIN;              //  1,048,576
  bf16raw* Abf = (bf16raw*)ws;                                    //  32 MB
  bf16raw* Wt  = (bf16raw*)(ws + nA * sizeof(bf16raw));           //   2 MB
  bf16raw* Uh  = (bf16raw*)(ws + (nA + nWt) * sizeof(bf16raw));   // 128 MB

  cvt_A <<<(int)((nA  + 255) / 256), 256, 0, stream>>>(u_vecs, Abf, (int)nA);
  cvt_Wt<<<(int)((nWt + 255) / 256), 256, 0, stream>>>(W, Wt);

  dim3 g(256, 32);   // 256 M-tile groups (128 rows each) x 32 capsules
  gemm_uhat<<<g, 256, 0, stream>>>(Abf, Wt, Uh);

  routing<<<B_BATCH, 512, 0, stream>>>(Uh, Out);
}
